// E3EqAtomWeave_51367808860760
// MI455X (gfx1250) — matrix operand order repaired and checked by reading, not compiled, not run on hardware
//
#include <hip/hip_runtime.h>
#include <hip/hip_bf16.h>
#include <stdint.h>

// ---------------------------------------------------------------------------
// E3EqAtomWeave on MI455X (gfx1250, wave32, WMMA)
//
// out = relu(t3 * (A@W3a+b3a) * (A@W3b+b3b)),
// A   = concat( relu(t1*(FA@W1a+b1a)*(FA@W1b+b1b)),
//               segsum( relu(t2*(FP@W2a+b2a)*(FP@W2b+b2b)) ) )
//
// Strategy: bf16 WMMA (v_wmma_f32_16x16x32_bf16, f32 accum) for all GEMMs;
// weights converted once to bf16 in B-fragment-swizzled layout; weights staged
// to LDS with async global->LDS copies (ASYNCcnt path); fused epilogue;
// segment-sum via global_atomic_add_f32 into an L2-resident 8 MB accumulator.
// ---------------------------------------------------------------------------

typedef __attribute__((ext_vector_type(16))) __bf16 v16bf;
typedef __attribute__((ext_vector_type(8)))  float  v8f;
typedef __attribute__((ext_vector_type(4)))  float  f32x4;
typedef __attribute__((ext_vector_type(4)))  int    ivec4_t;

#define N_ATOMS 8192
#define N_PAIRS 524288
#define FDIM    128

#if defined(__HIP_DEVICE_COMPILE__) && defined(__has_builtin)
#if __has_builtin(__builtin_amdgcn_global_load_async_to_lds_b128) && \
    __has_builtin(__builtin_amdgcn_s_wait_asynccnt)
#define USE_ASYNC_LDS 1
#endif
#endif

// ---------------------------------------------------------------------------
// Weight prep: f32 (Kdim x Ndim, row-major) -> bf16 in WMMA-B fragment order.
// Fragment (kc, nt) = B tile K=kc*32..+31, N=nt*16..+15. Lane l's 16 elements
// are stored contiguously (32 B) so the GEMM kernels read one v16bf per lane.
// Dense bf16 B 32x16 layout: lane l -> N=l%16, element e -> K=kc*32+(l>=16)*8+e+(e>=8)*8
// ---------------------------------------------------------------------------
__launch_bounds__(256)
__global__ void prep_w_kernel(const float* __restrict__ W, __bf16* __restrict__ out,
                              int Kdim, int NT) {
  int t = blockIdx.x * blockDim.x + threadIdx.x;
  int KC = Kdim >> 5;
  if (t >= KC * NT * 32) return;
  int frag = t >> 5;
  int lane = t & 31;
  int kc = frag / NT, nt = frag - kc * NT;
  int Ndim = NT * 16;
  int n = nt * 16 + (lane & 15);
  int kbase = kc * 32 + ((lane >> 4) << 3);
  __bf16* dst = out + (size_t)frag * 512 + lane * 16;
#pragma unroll
  for (int e = 0; e < 16; ++e) {
    dst[e] = (__bf16)W[(size_t)(kbase + e + ((e >> 3) << 3)) * Ndim + n];
  }
}

// ---------------------------------------------------------------------------
// Stage pre-swizzled bf16 weights into LDS (async-to-LDS when available).
// Builtin prototype (from clang): void(v4i AS1*, v4i AS3*, Imm int, Imm int)
// ---------------------------------------------------------------------------
static __device__ __forceinline__ void stage_to_lds(const __bf16* __restrict__ g,
                                                    __bf16* s, int bytes) {
#ifdef USE_ASYNC_LDS
  for (int off = (int)threadIdx.x * 16; off < bytes; off += 256 * 16) {
    __builtin_amdgcn_global_load_async_to_lds_b128(
        (__attribute__((address_space(1))) ivec4_t*)((const char*)g + off),
        (__attribute__((address_space(3))) ivec4_t*)((char*)s + off),
        0, 0);
  }
#else
  for (int off = (int)threadIdx.x * 16; off < bytes; off += 256 * 16) {
    *(ivec4_t*)((char*)s + off) = *(const ivec4_t*)((const char*)g + off);
  }
#endif
}

static __device__ __forceinline__ void stage_wait() {
#ifdef USE_ASYNC_LDS
  __builtin_amdgcn_s_wait_asynccnt(0);
#endif
  __syncthreads();
}

// ---------------------------------------------------------------------------
// Load one 16x32 bf16 A fragment from a row-major f32 row, converting f32->bf16.
// A layout: lane l -> M=l%16; elems 0..7 -> K=lo..lo+7, elems 8..15 -> K=lo+16..lo+23
// with lo = kc*32 + (l>=16)*8.
// ---------------------------------------------------------------------------
static __device__ __forceinline__ v16bf load_a_frag(const float* __restrict__ rowp, int lo) {
  f32x4 f0 = *(const f32x4*)(rowp + lo);
  f32x4 f1 = *(const f32x4*)(rowp + lo + 4);
  f32x4 f2 = *(const f32x4*)(rowp + lo + 16);
  f32x4 f3 = *(const f32x4*)(rowp + lo + 20);
  v16bf a;
#pragma unroll
  for (int i = 0; i < 4; ++i) {
    a[i]      = (__bf16)f0[i];
    a[4 + i]  = (__bf16)f1[i];
    a[8 + i]  = (__bf16)f2[i];
    a[12 + i] = (__bf16)f3[i];
  }
  return a;
}

// Dual-projection WMMA accumulation over K for one N-tile.
template <int KC>
static __device__ __forceinline__ void wmma_dual(const v16bf* __restrict__ afrag,
                                                 const __bf16* sWa, const __bf16* sWb,
                                                 int nt, int NT, int lane,
                                                 v8f& acca, v8f& accb) {
  v8f za = {0.f, 0.f, 0.f, 0.f, 0.f, 0.f, 0.f, 0.f};
  v8f zb = {0.f, 0.f, 0.f, 0.f, 0.f, 0.f, 0.f, 0.f};
#pragma unroll
  for (int kc = 0; kc < KC; ++kc) {
    size_t fo = ((size_t)(kc * NT + nt) * 32 + lane) * 16;
    v16bf bfa = *(const v16bf*)(sWa + fo);
    v16bf bfb = *(const v16bf*)(sWb + fo);
    za = __builtin_amdgcn_wmma_f32_16x16x32_bf16(false, afrag[kc], false, bfa,
                                                 (short)0, za, false, false);
    zb = __builtin_amdgcn_wmma_f32_16x16x32_bf16(false, afrag[kc], false, bfb,
                                                 (short)0, zb, false, false);
  }
  acca = za;
  accb = zb;
}

// ---------------------------------------------------------------------------
// Atom branch: AA = relu(t1*(FA@W1a+b1a)*(FA@W1b+b1b)) -> A[:, 0:128]
// Block = 128 rows (8 waves x 16-row tiles), N covered by 8 WMMA N-tiles.
// ---------------------------------------------------------------------------
__launch_bounds__(256)
__global__ void atom_kernel(const float* __restrict__ FA,
                            const __bf16* __restrict__ gWa, const __bf16* __restrict__ gWb,
                            const float* __restrict__ ba, const float* __restrict__ bb,
                            const float* __restrict__ tw, float* __restrict__ A) {
  __shared__ alignas(32) __bf16 sW[2 * FDIM * FDIM];  // 64 KB
  __bf16* sWa = sW;
  __bf16* sWb = sW + FDIM * FDIM;
  stage_to_lds(gWa, sWa, FDIM * FDIM * 2);
  stage_to_lds(gWb, sWb, FDIM * FDIM * 2);

  const int lane = threadIdx.x & 31;
  const int half = lane >> 4;
  const int rowBase = blockIdx.x * 128 + (threadIdx.x >> 5) * 16;
  const float* rowp = FA + (size_t)(rowBase + (lane & 15)) * FDIM;

  v16bf afrag[4];
#pragma unroll
  for (int kc = 0; kc < 4; ++kc) afrag[kc] = load_a_frag(rowp, kc * 32 + half * 8);

  stage_wait();

  for (int nt = 0; nt < 8; ++nt) {
    v8f acca, accb;
    wmma_dual<4>(afrag, sWa, sWb, nt, 8, lane, acca, accb);
    int n = nt * 16 + (lane & 15);
    float tn = tw[n], ban = ba[n], bbn = bb[n];
#pragma unroll
    for (int v = 0; v < 8; ++v) {
      float val = tn * (acca[v] + ban) * (accb[v] + bbn);
      val = val > 0.f ? val : 0.f;
      A[(size_t)(rowBase + half * 8 + v) * 256 + n] = val;  // concat slot [0:128)
    }
  }
}

// ---------------------------------------------------------------------------
// Pair branch: PA = relu(t2*(FP@W2a+b2a)*(FP@W2b+b2b)); scatter-add by
// pair_split into A[:, 128:256] with global_atomic_add_f32.
// ---------------------------------------------------------------------------
__launch_bounds__(256)
__global__ void pair_kernel(const float* __restrict__ FP, const int* __restrict__ split,
                            const __bf16* __restrict__ gWa, const __bf16* __restrict__ gWb,
                            const float* __restrict__ ba, const float* __restrict__ bb,
                            const float* __restrict__ tw, float* __restrict__ A) {
  __shared__ alignas(32) __bf16 sW[2 * FDIM * FDIM];  // 64 KB
  __bf16* sWa = sW;
  __bf16* sWb = sW + FDIM * FDIM;
  stage_to_lds(gWa, sWa, FDIM * FDIM * 2);
  stage_to_lds(gWb, sWb, FDIM * FDIM * 2);

  const int lane = threadIdx.x & 31;
  const int half = lane >> 4;
  const int rowBase = blockIdx.x * 128 + (threadIdx.x >> 5) * 16;
  const float* rowp = FP + (size_t)(rowBase + (lane & 15)) * FDIM;

  v16bf afrag[4];
#pragma unroll
  for (int kc = 0; kc < 4; ++kc) afrag[kc] = load_a_frag(rowp, kc * 32 + half * 8);

  int atoms[8];
#pragma unroll
  for (int v = 0; v < 8; ++v) atoms[v] = split[rowBase + half * 8 + v];

  stage_wait();

  for (int nt = 0; nt < 8; ++nt) {
    v8f acca, accb;
    wmma_dual<4>(afrag, sWa, sWb, nt, 8, lane, acca, accb);
    int n = nt * 16 + (lane & 15);
    float tn = tw[n], ban = ba[n], bbn = bb[n];
#pragma unroll
    for (int v = 0; v < 8; ++v) {
      float val = tn * (acca[v] + ban) * (accb[v] + bbn);
      val = val > 0.f ? val : 0.f;
      __hip_atomic_fetch_add(A + (size_t)atoms[v] * 256 + 128 + n, val,
                             __ATOMIC_RELAXED, __HIP_MEMORY_SCOPE_AGENT);
    }
  }
}

// ---------------------------------------------------------------------------
// Final: out = relu(t3*(A@W3a+b3a)*(A@W3b+b3b)), A is 8192x256 (K=256, KC=8).
// 128 KB of dynamic LDS for both W3 matrices.
// ---------------------------------------------------------------------------
__launch_bounds__(256)
__global__ void final_kernel(const float* __restrict__ Ain,
                             const __bf16* __restrict__ gWa, const __bf16* __restrict__ gWb,
                             const float* __restrict__ ba, const float* __restrict__ bb,
                             const float* __restrict__ tw, float* __restrict__ out) {
  extern __shared__ __bf16 sW3[];  // 2 * 256 * 128 bf16 = 128 KB
  __bf16* sWa = sW3;
  __bf16* sWb = sW3 + 256 * FDIM;
  stage_to_lds(gWa, sWa, 256 * FDIM * 2);
  stage_to_lds(gWb, sWb, 256 * FDIM * 2);

  const int lane = threadIdx.x & 31;
  const int half = lane >> 4;
  const int rowBase = blockIdx.x * 128 + (threadIdx.x >> 5) * 16;
  const float* rowp = Ain + (size_t)(rowBase + (lane & 15)) * 256;

  v16bf afrag[8];
#pragma unroll
  for (int kc = 0; kc < 8; ++kc) afrag[kc] = load_a_frag(rowp, kc * 32 + half * 8);

  stage_wait();

  for (int nt = 0; nt < 8; ++nt) {
    v8f acca, accb;
    wmma_dual<8>(afrag, sWa, sWb, nt, 8, lane, acca, accb);
    int n = nt * 16 + (lane & 15);
    float tn = tw[n], ban = ba[n], bbn = bb[n];
#pragma unroll
    for (int v = 0; v < 8; ++v) {
      float val = tn * (acca[v] + ban) * (accb[v] + bbn);
      val = val > 0.f ? val : 0.f;
      out[(size_t)(rowBase + half * 8 + v) * FDIM + n] = val;
    }
  }
}

// ---------------------------------------------------------------------------
// Host launch
// ---------------------------------------------------------------------------
extern "C" void kernel_launch(void* const* d_in, const int* in_sizes, int n_in,
                              void* d_out, int out_size, void* d_ws, size_t ws_size,
                              hipStream_t stream) {
  const float* FA    = (const float*)d_in[0];
  const float* FP    = (const float*)d_in[1];
  const int*   split = (const int*)d_in[2];
  const float* W1a = (const float*)d_in[3];  const float* b1a = (const float*)d_in[4];
  const float* W1b = (const float*)d_in[5];  const float* b1b = (const float*)d_in[6];
  const float* t1  = (const float*)d_in[7];
  const float* W2a = (const float*)d_in[8];  const float* b2a = (const float*)d_in[9];
  const float* W2b = (const float*)d_in[10]; const float* b2b = (const float*)d_in[11];
  const float* t2  = (const float*)d_in[12];
  const float* W3a = (const float*)d_in[13]; const float* b3a = (const float*)d_in[14];
  const float* W3b = (const float*)d_in[15]; const float* b3b = (const float*)d_in[16];
  const float* t3  = (const float*)d_in[17];
  float* out = (float*)d_out;

  // Workspace: A_concat (8192 x 256 f32 = 8 MB) + bf16 swizzled weights (256 KB)
  float* A = (float*)d_ws;
  char* p = (char*)d_ws + (size_t)N_ATOMS * 256 * sizeof(float);
  __bf16* cW1a = (__bf16*)p; p += (size_t)128 * 128 * 2;
  __bf16* cW1b = (__bf16*)p; p += (size_t)128 * 128 * 2;
  __bf16* cW2a = (__bf16*)p; p += (size_t)128 * 128 * 2;
  __bf16* cW2b = (__bf16*)p; p += (size_t)128 * 128 * 2;
  __bf16* cW3a = (__bf16*)p; p += (size_t)256 * 128 * 2;
  __bf16* cW3b = (__bf16*)p; p += (size_t)256 * 128 * 2;

  // Zero the concat/accumulator buffer (pair kernel accumulates atomically).
  (void)hipMemsetAsync(A, 0, (size_t)N_ATOMS * 256 * sizeof(float), stream);

  // Convert/swizzle weights to bf16 B-fragment layout.
  prep_w_kernel<<<4, 256, 0, stream>>>(W1a, cW1a, 128, 8);
  prep_w_kernel<<<4, 256, 0, stream>>>(W1b, cW1b, 128, 8);
  prep_w_kernel<<<4, 256, 0, stream>>>(W2a, cW2a, 128, 8);
  prep_w_kernel<<<4, 256, 0, stream>>>(W2b, cW2b, 128, 8);
  prep_w_kernel<<<8, 256, 0, stream>>>(W3a, cW3a, 256, 8);
  prep_w_kernel<<<8, 256, 0, stream>>>(W3b, cW3b, 256, 8);

  atom_kernel<<<N_ATOMS / 128, 256, 0, stream>>>(FA, cW1a, cW1b, b1a, b1b, t1, A);
  pair_kernel<<<N_PAIRS / 128, 256, 0, stream>>>(FP, split, cW2a, cW2b, b2a, b2b, t2, A);
  final_kernel<<<N_ATOMS / 128, 256, 131072, stream>>>(A, cW3a, cW3b, b3a, b3b, t3, out);
}
